// ClcHead_20332375179613
// MI455X (gfx1250) — compile-verified, matrix operand order checked by reading, not run on hardware
//
#include <hip/hip_runtime.h>
#include <hip/hip_bf16.h>

typedef __attribute__((ext_vector_type(16))) __bf16 v16bf;
typedef __attribute__((ext_vector_type(8)))  float  v8f;

#define B_   4
#define S_   2048
#define H_   768
#define SQ_  512            // queries per batch (rows with label > 0)
#define KT_  24             // 768 / 32  K-tiles
#define NT1_ 48             // 768 / 16  N-tiles (gemm1)
#define MT1_ 512            // 8192 / 16 M-tiles (gemm1)
#define MT2_ 32             // 512 / 16  M-tiles (gemm2)
#define NT2_ 128            // 2048 / 16 N-tiles (gemm2)

__device__ __forceinline__ unsigned short f2bf(float x) {
  unsigned int u = __float_as_uint(x);
  unsigned int r = u + 0x7fffu + ((u >> 16) & 1u);   // round-to-nearest-even
  return (unsigned short)(r >> 16);
}

// K position held by (lane, element) for 16-bit WMMA A/B fragments (ISA 7.12.2):
// lanes 0-15: K = {0..7, 16..23}; lanes 16-31: K = {8..15, 24..31}
__device__ __forceinline__ int kpat(int lane, int e) {
  int base = (lane < 16) ? 0 : 8;
  return base + e + ((e >= 8) ? 8 : 0);
}

// ---------------------------------------------------------------------------
// Kernel 1: segment-sum (exploits sorted indicator) -> bf16 A-fragments
// ---------------------------------------------------------------------------
__global__ void seg_sum_pack(const float* __restrict__ enc,
                             const int*   __restrict__ ind,
                             unsigned short* __restrict__ Afrag1) {
  __shared__ float rowbuf[H_];
  int row = blockIdx.x;                // 0..8191
  int b   = row / S_;
  int i   = row - b * S_;
  const int* indb = ind + b * S_;
  int v = indb[i];
  // lower bound
  int lo = 0, hi = S_;
  while (lo < hi) { int mid = (lo + hi) >> 1; if (indb[mid] <  v) lo = mid + 1; else hi = mid; }
  int seg_start = lo;
  // upper bound
  lo = 0; hi = S_;
  while (lo < hi) { int mid = (lo + hi) >> 1; if (indb[mid] <= v) lo = mid + 1; else hi = mid; }
  int seg_end = lo;

  const float* encb = enc + (size_t)b * S_ * H_;
  for (int h = threadIdx.x; h < H_; h += blockDim.x) {
    float s = 0.0f;
    for (int j = seg_start; j < seg_end; ++j) s += encb[(size_t)j * H_ + h];
    rowbuf[h] = s;
  }
  __syncthreads();

  int tm = row >> 4, ml = row & 15;
  for (int q = threadIdx.x; q < KT_ * 32; q += blockDim.x) {  // 768 packed slots
    int tk  = q >> 5;
    int rem = q & 31;
    int L   = ml + ((rem & 16) ? 16 : 0);
    int e   = rem & 15;
    int k   = tk * 32 + kpat(L, e);
    Afrag1[(((size_t)tm * KT_ + tk) * 32 + L) * 16 + e] = f2bf(rowbuf[k]);
  }
}

// ---------------------------------------------------------------------------
// Kernel 2: pack W^T into B-fragments   (B[k][n] = W[n][k])
// ---------------------------------------------------------------------------
__global__ void pack_W(const float* __restrict__ W,
                       unsigned short* __restrict__ BfragW) {
  int gid = blockIdx.x * blockDim.x + threadIdx.x;   // total 24*48*512
  int e  = gid & 15;
  int L  = (gid >> 4) & 31;
  int tn = (gid >> 9) % NT1_;
  int tk = gid / (512 * NT1_);
  int n  = tn * 16 + (L & 15);
  int k  = tk * 32 + kpat(L, e);
  BfragW[gid] = f2bf(W[(size_t)n * H_ + k]);
}

// ---------------------------------------------------------------------------
// Kernel 3: proj = tanh(agg @ W^T + b)  (one wave per 16x16 tile, bf16 WMMA)
// ---------------------------------------------------------------------------
__global__ void gemm1(const unsigned short* __restrict__ Afrag1,
                      const unsigned short* __restrict__ BfragW,
                      const float* __restrict__ bias,
                      unsigned short* __restrict__ proj) {
  int wave = threadIdx.x >> 5;
  int lane = threadIdx.x & 31;
  int tile = blockIdx.x * (blockDim.x >> 5) + wave;  // 0..24575
  int tm = tile / NT1_;
  int tn = tile - tm * NT1_;

  v8f acc = {};
  const unsigned short* aB = Afrag1 + (size_t)tm * KT_ * 512 + lane * 16;
  const unsigned short* bB = BfragW + (size_t)tn * 512 + lane * 16;
#pragma unroll
  for (int tk = 0; tk < KT_; ++tk) {
    v16bf a  = *reinterpret_cast<const v16bf*>(aB + (size_t)tk * 512);
    v16bf bm = *reinterpret_cast<const v16bf*>(bB + (size_t)tk * NT1_ * 512);
    acc = __builtin_amdgcn_wmma_f32_16x16x32_bf16(false, a, false, bm,
                                                  (short)0, acc, false, false);
  }
  int n   = tn * 16 + (lane & 15);
  float bv  = bias[n];
  int   mhi = (lane >= 16) ? 8 : 0;
#pragma unroll
  for (int r = 0; r < 8; ++r) {
    int m = tm * 16 + r + mhi;
    proj[(size_t)m * H_ + n] = f2bf(tanhf(acc[r] + bv));
  }
}

// ---------------------------------------------------------------------------
// Kernel 4: pack query rows (first 512 per batch) of proj into A-fragments
// ---------------------------------------------------------------------------
__global__ void pack_Q(const unsigned short* __restrict__ proj,
                       unsigned short* __restrict__ AfragQ) {
  int gid = blockIdx.x * blockDim.x + threadIdx.x;   // total 4*32*24*512
  int e   = gid & 15;
  int L   = (gid >> 4) & 31;
  int tk  = (gid >> 9) % KT_;
  int tmq = (gid / (512 * KT_)) % MT2_;
  int bb  = gid / (512 * KT_ * MT2_);
  int i = tmq * 16 + (L & 15);
  int k = tk * 32 + kpat(L, e);
  AfragQ[gid] = proj[((size_t)bb * S_ + i) * H_ + k];
}

// ---------------------------------------------------------------------------
// Kernel 5: pack proj^T (all 2048 keys per batch) into B-fragments
// ---------------------------------------------------------------------------
__global__ void pack_K(const unsigned short* __restrict__ proj,
                       unsigned short* __restrict__ BfragK) {
  int gid = blockIdx.x * blockDim.x + threadIdx.x;   // total 4*24*128*512
  int e  = gid & 15;
  int L  = (gid >> 4) & 31;
  int tn = (gid >> 9) % NT2_;
  int tk = (gid / (512 * NT2_)) % KT_;
  int bb = gid / (512 * NT2_ * KT_);
  int j = tn * 16 + (L & 15);
  int k = tk * 32 + kpat(L, e);
  BfragK[gid] = proj[((size_t)bb * S_ + j) * H_ + k];
}

// ---------------------------------------------------------------------------
// Kernel 6: scores = proj_q @ proj^T  per batch (bf16 WMMA, fp32 out)
// ---------------------------------------------------------------------------
__global__ void gemm2(const unsigned short* __restrict__ AfragQ,
                      const unsigned short* __restrict__ BfragK,
                      float* __restrict__ scores) {
  int wave = threadIdx.x >> 5;
  int lane = threadIdx.x & 31;
  int tile = blockIdx.x * (blockDim.x >> 5) + wave;  // 0..16383
  int tn   = tile % NT2_;
  int rest = tile / NT2_;
  int tmq  = rest % MT2_;
  int bb   = rest / MT2_;

  v8f acc = {};
  const unsigned short* aB = AfragQ + ((size_t)(bb * MT2_ + tmq) * KT_) * 512 + lane * 16;
  const unsigned short* bB = BfragK + ((size_t)bb * KT_ * NT2_ + tn) * 512 + lane * 16;
#pragma unroll
  for (int tk = 0; tk < KT_; ++tk) {
    v16bf a  = *reinterpret_cast<const v16bf*>(aB + (size_t)tk * 512);
    v16bf bm = *reinterpret_cast<const v16bf*>(bB + (size_t)tk * NT2_ * 512);
    acc = __builtin_amdgcn_wmma_f32_16x16x32_bf16(false, a, false, bm,
                                                  (short)0, acc, false, false);
  }
  int j   = tn * 16 + (lane & 15);
  int mhi = (lane >= 16) ? 8 : 0;
#pragma unroll
  for (int r = 0; r < 8; ++r) {
    int i = tmq * 16 + r + mhi;
    scores[((size_t)bb * SQ_ + i) * S_ + j] = acc[r];
  }
}

// ---------------------------------------------------------------------------
// Kernel 7: init the 6 outputs (losses/correct = 0, counts = N + 1e-6)
// ---------------------------------------------------------------------------
__global__ void init_out(float* out) {
  if (threadIdx.x == 0) {
    out[0] = 0.0f; out[1] = 0.0f; out[2] = 1024.0f + 1e-6f;
    out[3] = 0.0f; out[4] = 0.0f; out[5] = 1024.0f + 1e-6f;
  }
}

// ---------------------------------------------------------------------------
// Kernel 8: masked softmax-CE + argmax per query row, atomic reduce to 6 vals
// ---------------------------------------------------------------------------
__global__ void ce_reduce(const float* __restrict__ scores,
                          const int*   __restrict__ clc,
                          float* __restrict__ out) {
  __shared__ float smax[256];
  __shared__ int   sidx[256];
  __shared__ float ssum[256];
  int row = blockIdx.x;                 // 0..2047  (= b*512 + i)
  int bb  = row / SQ_;
  int i   = row - bb * SQ_;
  const float* sr = scores + (size_t)row * S_;
  const int*   cb = clc + bb * S_;
  int t = threadIdx.x;

  float mx = -3.4e38f; int mi = 0;
  for (int j = t; j < S_; j += 256) {
    float v = sr[j] + ((cb[j] < 0) ? 0.0f : -1000000000.0f);
    if (v > mx || (v == mx && j < mi)) { mx = v; mi = j; }
  }
  smax[t] = mx; sidx[t] = mi; __syncthreads();
  for (int s = 128; s > 0; s >>= 1) {
    if (t < s) {
      float o = smax[t + s]; int oi = sidx[t + s];
      if (o > smax[t] || (o == smax[t] && oi < sidx[t])) { smax[t] = o; sidx[t] = oi; }
    }
    __syncthreads();
  }
  float rmax = smax[0];
  int   amax = sidx[0];

  float se = 0.0f;
  for (int j = t; j < S_; j += 256) {
    float v = sr[j] + ((cb[j] < 0) ? 0.0f : -1000000000.0f);
    se += expf(v - rmax);
  }
  ssum[t] = se; __syncthreads();
  for (int s = 128; s > 0; s >>= 1) { if (t < s) ssum[t] += ssum[t + s]; __syncthreads(); }

  if (t == 0) {
    int lab = cb[i];                    // labels > 0 for these rows
    float vl   = sr[lab] + ((cb[lab] < 0) ? 0.0f : -1000000000.0f);
    float logp = (vl - rmax) - logf(ssum[0]);
    int g = (i & 1);                    // even i -> sep (g=0), odd -> tok (g=1)
    atomicAdd(&out[g * 3 + 0], -logp * (1.0f / 1024.0f));
    atomicAdd(&out[g * 3 + 1], (amax == lab) ? 1.0f : 0.0f);
  }
}

// ---------------------------------------------------------------------------
extern "C" void kernel_launch(void* const* d_in, const int* in_sizes, int n_in,
                              void* d_out, int out_size, void* d_ws, size_t ws_size,
                              hipStream_t stream) {
  const float* enc  = (const float*)d_in[0];   // (B,S,H) f32
  const float* W    = (const float*)d_in[1];   // (H,H)   f32
  const float* bias = (const float*)d_in[2];   // (H,)    f32
  const int*   ind  = (const int*)  d_in[3];   // (B,S)   i32 sorted
  const int*   clc  = (const int*)  d_in[4];   // (B,S)   i32
  float* out = (float*)d_out;

  char* ws = (char*)d_ws;
  size_t off = 0;
  unsigned short* Afrag1 = (unsigned short*)(ws + off); off += (size_t)MT1_ * KT_ * 512 * 2;         // 12.58 MB
  unsigned short* BfragW = (unsigned short*)(ws + off); off += (size_t)KT_ * NT1_ * 512 * 2;         //  1.18 MB
  unsigned short* proj   = (unsigned short*)(ws + off); off += (size_t)B_ * S_ * H_ * 2;             // 12.58 MB
  unsigned short* AfragQ = (unsigned short*)(ws + off); off += (size_t)B_ * MT2_ * KT_ * 512 * 2;    //  3.15 MB
  unsigned short* BfragK = (unsigned short*)(ws + off); off += (size_t)B_ * KT_ * NT2_ * 512 * 2;    // 12.58 MB
  float*          scores = (float*)(ws + off);          off += (size_t)B_ * SQ_ * S_ * 4;            // 16.78 MB

  init_out<<<1, 32, 0, stream>>>(out);
  seg_sum_pack<<<B_ * S_, 256, 0, stream>>>(enc, ind, Afrag1);
  pack_W<<<(KT_ * NT1_ * 512) / 256, 256, 0, stream>>>(W, BfragW);
  gemm1<<<(MT1_ * NT1_) / 8, 256, 0, stream>>>(Afrag1, BfragW, bias, proj);
  pack_Q<<<(B_ * MT2_ * KT_ * 512) / 256, 256, 0, stream>>>(proj, AfragQ);
  pack_K<<<(B_ * KT_ * NT2_ * 512) / 256, 256, 0, stream>>>(proj, BfragK);
  gemm2<<<(B_ * MT2_ * NT2_) / 8, 256, 0, stream>>>(AfragQ, BfragK, scores);
  ce_reduce<<<B_ * SQ_, 256, 0, stream>>>(scores, clc, out);
}
